// SClusterFormer_45122926411955
// MI455X (gfx1250) — matrix operand order repaired and checked by reading, not compiled, not run on hardware
//
#include <hip/hip_runtime.h>

// ---------------- problem constants ----------------
#define B_    64
#define P_    56
#define N_    (P_ * P_)          // 3136
#define DIM_  256
#define HEADS_ 4
#define HD_   24
#define HC_   (HEADS_ * HD_)     // 96
#define OUT_  256
#define M_    16                 // cluster centers (4x4)
#define BN_   (B_ * N_)          // 200704
#define FVC_  (2 * HC_)          // 192 fused feat|val columns

typedef _Float16 half_t;
typedef __attribute__((ext_vector_type(16))) _Float16 v16h;
typedef __attribute__((ext_vector_type(8)))  _Float16 h8;
typedef __attribute__((ext_vector_type(8)))  float    v8f;

// ---- A-fragment: 16x32 f16 (MxK) from row-major fp32, converted via v_cvt_pk_rtz ----
// lane<16 : row M=lane,    holds K = k0+{0..7, 16..23}
// lane>=16: row M=lane-16, holds K = k0+{8..15, 24..31}
static __device__ __forceinline__ v16h load_frag_a_f32(const float* __restrict__ src,
                                                       int ld, int k0, int lane) {
    const int half = lane >> 4, r = lane & 15;
    const float* row = src + (size_t)r * ld + k0 + 8 * half;
    float4 a0 = *(const float4*)(row);
    float4 a1 = *(const float4*)(row + 4);
    float4 b0 = *(const float4*)(row + 16);
    float4 b1 = *(const float4*)(row + 20);
    v16h out;
    {
        auto p0 = __builtin_amdgcn_cvt_pkrtz(a0.x, a0.y);
        auto p1 = __builtin_amdgcn_cvt_pkrtz(a0.z, a0.w);
        auto p2 = __builtin_amdgcn_cvt_pkrtz(a1.x, a1.y);
        auto p3 = __builtin_amdgcn_cvt_pkrtz(a1.z, a1.w);
        out[0] = p0[0]; out[1] = p0[1]; out[2] = p1[0]; out[3] = p1[1];
        out[4] = p2[0]; out[5] = p2[1]; out[6] = p3[0]; out[7] = p3[1];
    }
    {
        auto p0 = __builtin_amdgcn_cvt_pkrtz(b0.x, b0.y);
        auto p1 = __builtin_amdgcn_cvt_pkrtz(b0.z, b0.w);
        auto p2 = __builtin_amdgcn_cvt_pkrtz(b1.x, b1.y);
        auto p3 = __builtin_amdgcn_cvt_pkrtz(b1.z, b1.w);
        out[8]  = p0[0]; out[9]  = p0[1]; out[10] = p1[0]; out[11] = p1[1];
        out[12] = p2[0]; out[13] = p2[1]; out[14] = p3[0]; out[15] = p3[1];
    }
    return out;
}

// ---- B-fragment: 32x16 f16 (KxN), B[k][n] = Wh[colbase+n][k0+k], Wh pre-converted f16 ----
// lane<16 : col N=lane,    holds K = k0+{0..7, 16..23}
// lane>=16: col N=lane-16, holds K = k0+{8..15, 24..31}
static __device__ __forceinline__ v16h load_frag_b_f16(const half_t* __restrict__ wh,
                                                       int ld, int k0, int colbase, int lane) {
    const int half = lane >> 4, n = lane & 15;
    const half_t* row = wh + (size_t)(colbase + n) * ld + k0 + 8 * half;
    h8 lo = *(const h8*)(row);
    h8 hi = *(const h8*)(row + 16);
    v16h out;
#pragma unroll
    for (int j = 0; j < 8; ++j) { out[j] = lo[j]; out[8 + j] = hi[j]; }
    return out;
}

// ---------------- kernel 0: convert/fuse weights to f16 ----------------
__global__ __launch_bounds__(256) void prep_weights(const float* __restrict__ fw,
                                                    const float* __restrict__ vw,
                                                    const float* __restrict__ fb,
                                                    const float* __restrict__ vb,
                                                    const float* __restrict__ pw,
                                                    half_t* __restrict__ wfv,   // [192][256]
                                                    half_t* __restrict__ pwh,   // [256][96]
                                                    float* __restrict__ bfv) {  // [192]
    int i = (int)(blockIdx.x * blockDim.x + threadIdx.x);
    if (i < HC_ * DIM_)            wfv[i] = (half_t)fw[i];
    else if (i < FVC_ * DIM_)      wfv[i] = (half_t)vw[i - HC_ * DIM_];
    if (i < OUT_ * HC_)            pwh[i] = (half_t)pw[i];
    if (i < HC_)                   bfv[i] = fb[i];
    else if (i < FVC_)             bfv[i] = vb[i - HC_];
}

// ---------------- kernel 1: fused feat|val GEMM ----------------
// fv[BN_][192] = x[BN_][256] @ [f_w;v_w]^T + [f_b;v_b]
// Each wave: 1 row-tile x 4 col-tiles (one A fragment feeds 4 WMMAs per k-step)
__global__ __launch_bounds__(256) void fv_gemm(const float* __restrict__ x,
                                               const half_t* __restrict__ wfv,
                                               const float* __restrict__ bfv,
                                               float* __restrict__ fv) {
    const int wid  = (int)((blockIdx.x * blockDim.x + threadIdx.x) >> 5);
    const int lane = (int)(threadIdx.x & 31);
    const int NCG  = FVC_ / 64;          // 3 column groups of 4 tiles
    const int tr = wid / NCG;
    const int cg = wid % NCG;
    if (tr >= BN_ / 16) return;          // wave-uniform guard

    const int row0 = tr * 16;
    const int col0 = cg * 64;

    v8f c0 = {0.f,0.f,0.f,0.f,0.f,0.f,0.f,0.f};
    v8f c1 = c0, c2 = c0, c3 = c0;

#pragma unroll
    for (int k0 = 0; k0 < DIM_; k0 += 32) {
        v16h a  = load_frag_a_f32(x + (size_t)row0 * DIM_, DIM_, k0, lane);
        v16h b0 = load_frag_b_f16(wfv, DIM_, k0, col0,      lane);
        v16h b1 = load_frag_b_f16(wfv, DIM_, k0, col0 + 16, lane);
        v16h b2 = load_frag_b_f16(wfv, DIM_, k0, col0 + 32, lane);
        v16h b3 = load_frag_b_f16(wfv, DIM_, k0, col0 + 48, lane);
        c0 = __builtin_amdgcn_wmma_f32_16x16x32_f16(false, a, false, b0, (short)0, c0, false, false);
        c1 = __builtin_amdgcn_wmma_f32_16x16x32_f16(false, a, false, b1, (short)0, c1, false, false);
        c2 = __builtin_amdgcn_wmma_f32_16x16x32_f16(false, a, false, b2, (short)0, c2, false, false);
        c3 = __builtin_amdgcn_wmma_f32_16x16x32_f16(false, a, false, b3, (short)0, c3, false, false);
    }

    const int half = lane >> 4, n = lane & 15;
#pragma unroll
    for (int t = 0; t < 4; ++t) {
        v8f c = (t == 0) ? c0 : (t == 1) ? c1 : (t == 2) ? c2 : c3;
        int col = col0 + t * 16 + n;
        float bv = bfv[col];
#pragma unroll
        for (int i = 0; i < 8; ++i) {
            int row = row0 + 8 * half + i;
            fv[(size_t)row * FVC_ + col] = c[i] + bv;
        }
    }
}

// ---------------- kernel 2: clustering (one block per b*head) ----------------
__global__ __launch_bounds__(256) void cluster_kernel(const float* __restrict__ fv,
                                                      float* __restrict__ merged,
                                                      const float* __restrict__ alpha_p,
                                                      const float* __restrict__ beta_p) {
    __shared__ float s_cent[M_][HD_];
    __shared__ float s_vcent[M_][HD_];
    __shared__ float s_cn[M_][HD_];
    __shared__ float s_agg[M_][HD_];
    __shared__ float s_cnt[M_];
    __shared__ float s_simval[N_];
    __shared__ unsigned char s_assign[N_];

    const int tid = (int)threadIdx.x;
    const int bh  = (int)blockIdx.x;
    const int b   = bh / HEADS_;
    const int h   = bh % HEADS_;
    const float alpha = alpha_p[0];
    const float beta  = beta_p[0];
    const float* fvb  = fv + (size_t)b * N_ * FVC_;

    for (int i = tid; i < M_ * HD_; i += 256) {
        (&s_cent[0][0])[i]  = 0.f;
        (&s_vcent[0][0])[i] = 0.f;
        (&s_agg[0][0])[i]   = 0.f;
    }
    if (tid < M_) s_cnt[tid] = 0.f;
    __syncthreads();

    // pool: block means over 14x14 patches (w-major token order n = w*56 + h)
    for (int n = tid; n < N_; n += 256) {
        int w = n / P_, hh = n % P_;
        int m = (w / 14) * 4 + (hh / 14);
        const float* fr = fvb + (size_t)n * FVC_ + h * HD_;
        const float* vr = fr + HC_;
#pragma unroll
        for (int d = 0; d < HD_; ++d) {
            atomicAdd(&s_cent[m][d],  fr[d]);
            atomicAdd(&s_vcent[m][d], vr[d]);
        }
    }
    __syncthreads();

    if (tid < M_) {
        float nrm = 0.f;
#pragma unroll
        for (int d = 0; d < HD_; ++d) {
            float c = s_cent[tid][d] * (1.0f / 196.0f);
            s_cent[tid][d] = c;
            nrm += c * c;
        }
        float inv = 1.0f / fmaxf(sqrtf(nrm), 1e-12f);
#pragma unroll
        for (int d = 0; d < HD_; ++d) {
            s_cn[tid][d] = s_cent[tid][d] * inv;
            s_vcent[tid][d] *= (1.0f / 196.0f);
        }
    }
    __syncthreads();

    // per-token: cosine sim vs 16 centers, leaky_relu, top-1, aggregate
    for (int n = tid; n < N_; n += 256) {
        const float* fr = fvb + (size_t)n * FVC_ + h * HD_;
        float fl[HD_];
        float nrm = 0.f;
#pragma unroll
        for (int d = 0; d < HD_; ++d) { fl[d] = fr[d]; nrm += fl[d] * fl[d]; }
        float inv = 1.0f / fmaxf(sqrtf(nrm), 1e-12f);

        float best = -3.4e38f;
        int bm = 0;
#pragma unroll
        for (int m = 0; m < M_; ++m) {
            float s = 0.f;
#pragma unroll
            for (int d = 0; d < HD_; ++d) s += s_cn[m][d] * fl[d];
            s *= inv;
            float t = beta + alpha * s;
            t = (t > 0.f) ? t : 0.2f * t;
            if (t > best) { best = t; bm = m; }
        }
        s_simval[n] = best;
        s_assign[n] = (unsigned char)bm;
        atomicAdd(&s_cnt[bm], 1.0f);
        const float* vr = fr + HC_;
#pragma unroll
        for (int d = 0; d < HD_; ++d) atomicAdd(&s_agg[bm][d], best * vr[d]);
    }
    __syncthreads();

    if (tid < M_) {
        float inv = 1.0f / (s_cnt[tid] + 1.0f);
#pragma unroll
        for (int d = 0; d < HD_; ++d)
            s_agg[tid][d] = (s_agg[tid][d] + s_vcent[tid][d]) * inv;
    }
    __syncthreads();

    // dispatch back
    float* mb = merged + (size_t)b * N_ * HC_ + h * HD_;
    for (int n = tid; n < N_; n += 256) {
        int m = s_assign[n];
        float sv = s_simval[n];
#pragma unroll
        for (int d = 0; d < HD_; ++d)
            mb[(size_t)n * HC_ + d] = sv * s_agg[m][d];
    }
}

// ---------------- kernel 3: output projection GEMM ----------------
// out[BN_][256] = merged[BN_][96] @ proj_w^T + proj_b ; 1 row-tile x 4 col-tiles / wave
__global__ __launch_bounds__(256) void proj_gemm(const float* __restrict__ merged,
                                                 const half_t* __restrict__ pwh,
                                                 const float* __restrict__ pb,
                                                 float* __restrict__ out) {
    const int wid  = (int)((blockIdx.x * blockDim.x + threadIdx.x) >> 5);
    const int lane = (int)(threadIdx.x & 31);
    const int NCG  = OUT_ / 64;          // 4 column groups of 4 tiles
    const int tr = wid / NCG;
    const int cg = wid % NCG;
    if (tr >= BN_ / 16) return;          // wave-uniform guard

    const int row0 = tr * 16;
    const int col0 = cg * 64;

    v8f c0 = {0.f,0.f,0.f,0.f,0.f,0.f,0.f,0.f};
    v8f c1 = c0, c2 = c0, c3 = c0;

#pragma unroll
    for (int k0 = 0; k0 < HC_; k0 += 32) {
        v16h a  = load_frag_a_f32(merged + (size_t)row0 * HC_, HC_, k0, lane);
        v16h b0 = load_frag_b_f16(pwh, HC_, k0, col0,      lane);
        v16h b1 = load_frag_b_f16(pwh, HC_, k0, col0 + 16, lane);
        v16h b2 = load_frag_b_f16(pwh, HC_, k0, col0 + 32, lane);
        v16h b3 = load_frag_b_f16(pwh, HC_, k0, col0 + 48, lane);
        c0 = __builtin_amdgcn_wmma_f32_16x16x32_f16(false, a, false, b0, (short)0, c0, false, false);
        c1 = __builtin_amdgcn_wmma_f32_16x16x32_f16(false, a, false, b1, (short)0, c1, false, false);
        c2 = __builtin_amdgcn_wmma_f32_16x16x32_f16(false, a, false, b2, (short)0, c2, false, false);
        c3 = __builtin_amdgcn_wmma_f32_16x16x32_f16(false, a, false, b3, (short)0, c3, false, false);
    }

    const int half = lane >> 4, n = lane & 15;
#pragma unroll
    for (int t = 0; t < 4; ++t) {
        v8f c = (t == 0) ? c0 : (t == 1) ? c1 : (t == 2) ? c2 : c3;
        int col = col0 + t * 16 + n;
        float bv = pb[col];
#pragma unroll
        for (int i = 0; i < 8; ++i) {
            int row = row0 + 8 * half + i;
            out[(size_t)row * OUT_ + col] = c[i] + bv;
        }
    }
}

// ---------------- launcher ----------------
extern "C" void kernel_launch(void* const* d_in, const int* in_sizes, int n_in,
                              void* d_out, int out_size, void* d_ws, size_t ws_size,
                              hipStream_t stream) {
    (void)in_sizes; (void)n_in; (void)out_size; (void)ws_size;

    const float* x   = (const float*)d_in[0];
    const float* fw  = (const float*)d_in[1];
    const float* fb  = (const float*)d_in[2];
    const float* vw  = (const float*)d_in[3];
    const float* vb  = (const float*)d_in[4];
    const float* pw  = (const float*)d_in[5];
    const float* pb  = (const float*)d_in[6];
    const float* sa  = (const float*)d_in[7];
    const float* sb  = (const float*)d_in[8];
    float* out = (float*)d_out;

    // workspace layout (floats, then f16 tails)
    float* fv     = (float*)d_ws;                            // [BN_ x 192]
    float* merged = fv + (size_t)BN_ * FVC_;                 // [BN_ x 96]
    float* bfv    = merged + (size_t)BN_ * HC_;              // [192]
    half_t* wfv   = (half_t*)(bfv + FVC_);                   // [192 x 256] f16
    half_t* pwh   = wfv + (size_t)FVC_ * DIM_;               // [256 x 96]  f16

    // kernel 0: weight conversion/fusion (tiny)
    prep_weights<<<(FVC_ * DIM_ + 255) / 256, 256, 0, stream>>>(fw, vw, fb, vb, pw,
                                                                wfv, pwh, bfv);
    // kernel 1: (BN/16) row-tiles x 3 col-groups = 37632 waves, 8 waves/block
    {
        int waves = (BN_ / 16) * (FVC_ / 64);
        fv_gemm<<<waves / 8, 256, 0, stream>>>(x, wfv, bfv, fv);
    }
    // kernel 2: one block per (batch, head)
    cluster_kernel<<<B_ * HEADS_, 256, 0, stream>>>(fv, merged, sa, sb);
    // kernel 3: (BN/16) row-tiles x 4 col-groups = 50176 waves, 8 waves/block
    {
        int waves = (BN_ / 16) * (OUT_ / 64);
        proj_gemm<<<waves / 8, 256, 0, stream>>>(merged, pwh, pb, out);
    }
}
